// Head_51651276702474
// MI455X (gfx1250) — matrix operand order repaired and checked by reading, not compiled, not run on hardware
//
#include <hip/hip_runtime.h>

typedef __attribute__((ext_vector_type(16))) __bf16        v16bf;
typedef __attribute__((ext_vector_type(2)))  __bf16        bf2x;
typedef __attribute__((ext_vector_type(8)))  float         v8f;
typedef __attribute__((ext_vector_type(8)))  unsigned int  v8u;
typedef __attribute__((ext_vector_type(2)))  float         fx2;

#define T_LEN   4096
#define NEMBD   512
#define DIM     64

// LDS row strides (uints): padded for bank-conflict-free fragment loads,
// 16B-aligned rows so b128 ops stay legal.
#define SK_STRIDE   36   // K tile row-major:  32 rows x (32+4) uints  (144 B)
#define SKT_STRIDE  20   // K tile dim-major:  64 rows x (16+4) uints  ( 80 B)
#define SP_STRIDE   20   // P tile row-major:  16 rows x (16+4) uints  ( 80 B)

__device__ __forceinline__ unsigned short f2bf(float f) {
    return __builtin_bit_cast(unsigned short, (__bf16)f);
}
// packed pair -> single v_cvt_pk_bf16_f32
__device__ __forceinline__ unsigned int f2bf2(float lo, float hi) {
    fx2 f; f.x = lo; f.y = hi;
    bf2x h = __builtin_convertvector(f, bf2x);
    return __builtin_bit_cast(unsigned int, h);
}
__device__ __forceinline__ v16bf asbf16(v8u u) {
    return __builtin_bit_cast(v16bf, u);
}

// A-matrix (16-bit, 16x32) K-pair base for VGPR v, lane-group g (contiguous k = 16g + 2v, matches B)
__device__ __forceinline__ int kpairA(int v, int g) {
    return (v << 1) + (g << 4);
}
// B-matrix (16-bit, 32x16) K-pair base for VGPR v, lane-group g
__device__ __forceinline__ int kpairB(int v, int g) {
    return (v << 1) + (g << 4);
}

__device__ __forceinline__ v8f wmma_bf16(v16bf a, v16bf b, v8f c) {
    return __builtin_amdgcn_wmma_f32_16x16x32_bf16(
        false, a, false, b, (short)0, c, false, false);
}

// --- CDNA5 async LDS copy: GLOBAL_LOAD_ASYNC_TO_LDS_B128 (GVS form) --------
__device__ __forceinline__ void async_copy_b128(unsigned lds_addr,
                                                const void* gbase,
                                                unsigned voff) {
    asm volatile("global_load_async_to_lds_b128 %0, %1, %2"
                 :
                 : "v"(lds_addr), "v"(voff), "s"(gbase)
                 : "memory");
}

__device__ __forceinline__ void wait_async0() {
#if __has_builtin(__builtin_amdgcn_s_wait_asynccnt)
    __builtin_amdgcn_s_wait_asynccnt(0);
#else
    asm volatile("s_wait_asynccnt 0x0" ::: "memory");
#endif
}

// ---------------------------------------------------------------------------
// Kernel 0: weights fp32 [512,64] -> bf16 transposed [64,512]
// ---------------------------------------------------------------------------
__global__ __launch_bounds__(256) void prep_w_kernel(
    const float* __restrict__ wq, const float* __restrict__ wk,
    unsigned short* __restrict__ wtq, unsigned short* __restrict__ wtk)
{
    int i = blockIdx.x * blockDim.x + threadIdx.x;
    if (i < NEMBD * DIM) {
        int k = i >> 6;
        int n = i & 63;
        wtq[n * NEMBD + k] = f2bf(wq[i]);
        wtk[n * NEMBD + k] = f2bf(wk[i]);
    }
}

// ---------------------------------------------------------------------------
// Kernel 1: Q = x@Wq, K = x@Wk as bf16 (A fragment reused for 8 WMMAs/chunk)
// ---------------------------------------------------------------------------
__global__ __launch_bounds__(128) void proj_kernel(
    const float* __restrict__ x,
    const unsigned short* __restrict__ wtq,
    const unsigned short* __restrict__ wtk,
    unsigned short* __restrict__ qbf,
    unsigned short* __restrict__ kbf)
{
    const int lane = threadIdx.x & 31;
    const int wave = threadIdx.x >> 5;
    const int mn   = lane & 15;
    const int g    = lane >> 4;
    const int rowbase = (blockIdx.x * 4 + wave) * 16;

    const unsigned int* wtqU = (const unsigned int*)wtq;
    const unsigned int* wtkU = (const unsigned int*)wtk;

    const v8f vzero = {0.f, 0.f, 0.f, 0.f, 0.f, 0.f, 0.f, 0.f};
    v8f accq[4] = {vzero, vzero, vzero, vzero};
    v8f acck[4] = {vzero, vzero, vzero, vzero};

    for (int kc = 0; kc < NEMBD / 32; ++kc) {
        v8u au;
        const float* xrow = x + (size_t)(rowbase + mn) * NEMBD + kc * 32;
        __builtin_prefetch(xrow + 256, 0, 0);
#pragma unroll
        for (int v = 0; v < 8; ++v) {
            int k0 = kpairA(v, g);
            fx2 p = __builtin_nontemporal_load((const fx2*)(xrow + k0));
            au[v] = f2bf2(p.x, p.y);
        }
        v16bf a = asbf16(au);
#pragma unroll
        for (int t = 0; t < 4; ++t) {
            int col = t * 16 + mn;
            v8u bq, bk;
#pragma unroll
            for (int v = 0; v < 8; ++v) {
                int k0 = kpairB(v, g);
                unsigned int idx = col * (NEMBD / 2) + kc * 16 + (k0 >> 1);
                bq[v] = wtqU[idx];
                bk[v] = wtkU[idx];
            }
            accq[t] = wmma_bf16(a, asbf16(bq), accq[t]);
            acck[t] = wmma_bf16(a, asbf16(bk), acck[t]);
        }
    }
#pragma unroll
    for (int t = 0; t < 4; ++t)
#pragma unroll
        for (int v = 0; v < 8; ++v) {
            int r = v + 8 * g;
            size_t idx = (size_t)(rowbase + r) * DIM + t * 16 + mn;
            qbf[idx] = f2bf(accq[t][v]);
            kbf[idx] = f2bf(acck[t][v]);
        }
}

// ---------------------------------------------------------------------------
// Kernel 2: causal flash attention, V == K (reference bug).
// One workgroup = (batch, 64-row query tile), 4 waves x 16 rows.
// 32-key blocks double-buffered in LDS via async-to-LDS B128 copies;
// per-block LDS transpose gives contiguous PV fragments.
// ---------------------------------------------------------------------------
__global__ __launch_bounds__(128) void attn_kernel(
    const unsigned short* __restrict__ qbf,
    const unsigned short* __restrict__ kbf,
    float* __restrict__ out)
{
    __shared__ __align__(16) unsigned int sK [2][32 * SK_STRIDE];  // row-major tile
    __shared__ __align__(16) unsigned int sKt[2][64 * SKT_STRIDE]; // dim-major tile
    __shared__ __align__(16) unsigned int sP [4][16 * SP_STRIDE];  // per-wave P

    const int tid  = threadIdx.x;
    const int lane = tid & 31;
    const int wave = tid >> 5;
    const int mn   = lane & 15;
    const int g    = lane >> 4;

    const int b  = blockIdx.x >> 6;
    const int q0 = (blockIdx.x & 63) * 64;
    const int qbase = q0 + wave * 16;
    const size_t rowb = (size_t)b * T_LEN;

    const unsigned int* qbfU = (const unsigned int*)qbf;
    unsigned short* sPu = (unsigned short*)&sP[wave][0];
    const unsigned int* sPw = &sP[wave][0];

    // Preload Q fragments: two 16x32 bf16 chunks of this wave's 16 rows
    v16bf aq[2];
#pragma unroll
    for (int c = 0; c < 2; ++c) {
        v8u u;
#pragma unroll
        for (int v = 0; v < 8; ++v) {
            int k0 = kpairA(v, g);
            u[v] = qbfU[(rowb + qbase + mn) * (DIM / 2) + c * 16 + (k0 >> 1)];
        }
        aq[c] = asbf16(u);
    }

    const v8f vzero = {0.f, 0.f, 0.f, 0.f, 0.f, 0.f, 0.f, 0.f};
    v8f accO[4] = {vzero, vzero, vzero, vzero};
    float mrow[8], lrow[8];
#pragma unroll
    for (int v = 0; v < 8; ++v) { mrow[v] = -1e30f; lrow[v] = 0.f; }

    const int nb = (q0 >> 5) + 2;          // key blocks covering keys <= q0+63
    const float scale = 0.125f;            // 1/sqrt(64)

    // async fill of one tile: 32 rows x 128 B -> padded LDS rows of 144 B
    auto issueTile = [&](int kbi, int buf) {
        const char* gb = (const char*)kbf + (rowb + (size_t)kbi * 32) * (DIM * 2);
        unsigned lds0 = (unsigned)(size_t)&sK[buf][0];
        int chunk = tid & 7;               // 8 x 16B per row
#pragma unroll
        for (int j = 0; j < 2; ++j) {
            int row = (tid >> 3) + 16 * j;
            async_copy_b128(lds0 + row * (SK_STRIDE * 4) + chunk * 16,
                            gb, (unsigned)(j * 2048 + tid * 16));
        }
    };

    issueTile(0, 0);
    for (int kb = 0; kb < nb; ++kb) {
        const int buf  = kb & 1;
        const int key0 = kb * 32;
        wait_async0();
        __syncthreads();                    // tile[buf] visible to all waves
        if (kb + 1 < nb) issueTile(kb + 1, buf ^ 1);   // overlap next copy

        const unsigned int* sKb = &sK[buf][0];

        // --- build dim-major copy for PV fragments (16 u16 stores/thread) ---
        {
            unsigned short* sKtu = (unsigned short*)&sKt[buf][0];
            int row = tid & 31;                      // key
#pragma unroll
            for (int j = 0; j < 8; ++j) {
                int d2 = (tid >> 5) * 8 + j;         // dim pair
                unsigned w = sKb[row * SK_STRIDE + d2];
                sKtu[(2 * d2)     * (SKT_STRIDE * 2) + row] = (unsigned short)w;
                sKtu[(2 * d2 + 1) * (SKT_STRIDE * 2) + row] = (unsigned short)(w >> 16);
            }
        }
        __syncthreads();                    // sKt[buf] visible

        // --- S = Q K^T : two 16x16 score tiles, K-dim 64 = 2 chunks ---
        v8f s0 = vzero, s1 = vzero;
#pragma unroll
        for (int c = 0; c < 2; ++c) {
            v8u b0, b1;
#pragma unroll
            for (int v = 0; v < 8; ++v) {
                b0[v] = sKb[(mn)      * SK_STRIDE + c * 16 + v + 8 * g];
                b1[v] = sKb[(16 + mn) * SK_STRIDE + c * 16 + v + 8 * g];
            }
            s0 = wmma_bf16(aq[c], asbf16(b0), s0);
            s1 = wmma_bf16(aq[c], asbf16(b1), s1);
        }

        // --- online softmax (row stats live in 16-lane half-waves) ---
        float e0[8], e1[8];
#pragma unroll
        for (int v = 0; v < 8; ++v) {
            int qrow = qbase + v + 8 * g;
            float a0 = s0[v] * scale; if (key0 + mn      > qrow) a0 = -1e30f;
            float a1 = s1[v] * scale; if (key0 + 16 + mn > qrow) a1 = -1e30f;
            float mb = fmaxf(a0, a1);
#pragma unroll
            for (int d = 1; d < 16; d <<= 1)
                mb = fmaxf(mb, __shfl_xor(mb, d, 32));
            float mnew = fmaxf(mrow[v], mb);
            float al = __expf(mrow[v] - mnew);
            float p0 = __expf(a0 - mnew);
            float p1 = __expf(a1 - mnew);
            float rs = p0 + p1;
#pragma unroll
            for (int d = 1; d < 16; d <<= 1)
                rs += __shfl_xor(rs, d, 32);
            lrow[v] = lrow[v] * al + rs;
            mrow[v] = mnew;
#pragma unroll
            for (int t = 0; t < 4; ++t) accO[t][v] *= al;
            e0[v] = p0; e1[v] = p1;
        }

        // --- P (C layout) -> per-wave LDS 16x32 bf16 -> A fragment ---
#pragma unroll
        for (int v = 0; v < 8; ++v) {
            int r = v + 8 * g;
            sPu[r * (SP_STRIDE * 2) + mn]      = f2bf(e0[v]);
            sPu[r * (SP_STRIDE * 2) + 16 + mn] = f2bf(e1[v]);
        }
        v8u pu;
#pragma unroll
        for (int v = 0; v < 8; ++v) {
            int k0 = kpairA(v, g);
            pu[v] = sPw[mn * SP_STRIDE + (k0 >> 1)];
        }
        v16bf ap = asbf16(pu);

        // --- O += P * V : contiguous b128-able fragments from sKt ---
        {
            const unsigned int* sKtU = &sKt[buf][0];
#pragma unroll
            for (int t = 0; t < 4; ++t) {
                v8u vu;
#pragma unroll
                for (int v = 0; v < 8; ++v)
                    vu[v] = sKtU[(t * 16 + mn) * SKT_STRIDE + v + 8 * g];
                accO[t] = wmma_bf16(ap, asbf16(vu), accO[t]);
            }
        }
    }

    // --- epilogue: normalize and store fp32 (streamed once -> NT stores) ---
#pragma unroll
    for (int t = 0; t < 4; ++t)
#pragma unroll
        for (int v = 0; v < 8; ++v) {
            int r = v + 8 * g;
            float inv = 1.0f / lrow[v];
            __builtin_nontemporal_store(accO[t][v] * inv,
                out + (rowb + qbase + r) * DIM + t * 16 + mn);
        }
}

// ---------------------------------------------------------------------------
extern "C" void kernel_launch(void* const* d_in, const int* in_sizes, int n_in,
                              void* d_out, int out_size, void* d_ws, size_t ws_size,
                              hipStream_t stream) {
    (void)in_sizes; (void)n_in; (void)out_size; (void)ws_size;
    const float* x  = (const float*)d_in[0];
    const float* wq = (const float*)d_in[1];
    const float* wk = (const float*)d_in[2];
    // d_in[3] (w_v) intentionally unused: reference uses w_k for v (bug kept)
    float* out = (float*)d_out;

    unsigned short* wtq = (unsigned short*)d_ws;            // 64*512 bf16
    unsigned short* wtk = wtq + DIM * NEMBD;                // 64*512 bf16
    unsigned short* qbf = wtk + DIM * NEMBD;                // 32768*64 bf16
    unsigned short* kbf = qbf + (size_t)8 * T_LEN * DIM;    // 32768*64 bf16

    prep_w_kernel<<<(NEMBD * DIM + 255) / 256, 256, 0, stream>>>(wq, wk, wtq, wtk);
    proj_kernel<<<(8 * T_LEN) / 64, 128, 0, stream>>>(x, wtq, wtk, qbf, kbf);
    attn_kernel<<<8 * (T_LEN / 64), 128, 0, stream>>>(qbf, kbf, out);
}
